// GenCastCore_12781822673003
// MI455X (gfx1250) — matrix operand order repaired and checked by reading, not compiled, not run on hardware
//
#include <hip/hip_runtime.h>
#include <hip/hip_bf16.h>
#include <math.h>

typedef __attribute__((ext_vector_type(16))) __bf16 v16bf;
typedef __attribute__((ext_vector_type(8)))  float  v8f;

#define NGRID 64800
#define NMESH 10242
#define NEDGE 150000
#define ECHUNK 25000
#define GINF 109
#define GLD 112            // padded leading dim for grid features (16B aligned rows)

// A-operand descriptor: virtual concat of up to 3 sources.
struct AGather {
  const float* src0; const int* idx0; int ld0; int s0;
  const float* src1; const int* idx1; int ld1; int s01;
  const float* src2; int ld2;
};

struct BF16Frag { uint4 lo, hi; };

__device__ __forceinline__ void stage16_vec(const float* __restrict__ p, __bf16* dst) {
  const float4* q = (const float4*)p;
  float4 x[4];
  #pragma unroll
  for (int u = 0; u < 4; ++u) x[u] = q[u];
  #pragma unroll
  for (int u = 0; u < 4; ++u) {
    dst[4 * u + 0] = (__bf16)x[u].x;
    dst[4 * u + 1] = (__bf16)x[u].y;
    dst[4 * u + 2] = (__bf16)x[u].z;
    dst[4 * u + 3] = (__bf16)x[u].w;
  }
}

__device__ __forceinline__ void stage16_zero(__bf16* dst) {
  #pragma unroll
  for (int j = 0; j < 16; ++j) dst[j] = (__bf16)0.f;
}

// ---------------------------------------------------------------------------
// WMMA GEMM: C[M,N] = epilogue(A[M,K] @ W[K,N] + bias), W pre-converted bf16.
// 256 threads = 8 waves; block tile 128x64; each wave a 32x32 sub-tile
// -> 4x v_wmma_f32_16x16x32_bf16 per K step with fragment reuse.
// AMODE: 0 plain, 1 edge concat [gather|gather|efeat], 2 concat [x|x] no idx
// mode : 0 store, 1 add (residual), 2 atomic scatter-add via scat[row]
// act  : 0 none, 1 SiLU
// ---------------------------------------------------------------------------
template<int AMODE>
__global__ __launch_bounds__(256) void wmma_gemm(
    AGather A, int M, int Kdim, const __bf16* __restrict__ W,
    const float* __restrict__ bias, int N, float* __restrict__ C,
    const int* __restrict__ scat, int mode, int act)
{
  __shared__ __align__(16) __bf16 lds_a[128][32];
  __shared__ __align__(16) __bf16 lds_b[64][32];   // transposed: [n][k]
  const int tid  = threadIdx.x;
  const int lane = tid & 31;
  const int wave = tid >> 5;
  const int wm = wave & 3;            // 4 row groups of 32
  const int wn = wave >> 2;           // 2 col groups of 32
  const int rowBlk = blockIdx.x * 128;
  const int colBlk = blockIdx.y * 64;

  // ---- per-thread A staging coords: 2 threads/row, 16 k each ----
  const int ar   = tid >> 1;          // 0..127
  const int aks  = (tid & 1) * 16;    // 0 or 16
  const int agrow = rowBlk + ar;
  const bool arow_ok = agrow < M;
  const int asafe = arow_ok ? agrow : 0;

  const float* ap0 = nullptr; const float* ap1 = nullptr; const float* ap2 = nullptr;
  if (AMODE == 0) {
    ap0 = A.src0 + (size_t)asafe * A.ld0;
  } else if (AMODE == 1) {
    int s  = A.idx0[asafe];            // gather indices resolved ONCE
    int rr = A.idx1[asafe];
    ap0 = A.src0 + (size_t)s  * A.ld0;
    ap1 = A.src1 + (size_t)rr * A.ld1;
    ap2 = A.src2 + (size_t)asafe * A.ld2;
  } else {
    ap0 = A.src0 + (size_t)asafe * A.ld0;
    ap1 = A.src1 + (size_t)asafe * A.ld1;
  }
  const bool a_vec_ok = (AMODE != 0) || ((A.ld0 & 3) == 0);

  // ---- per-thread B staging coords: n = tid&63, 8 k each ----
  const int bn  = tid & 63;
  const int bk0 = (tid >> 6) * 8;
  const int bcol_g = colBlk + bn;
  const bool bcol_ok = bcol_g < N;

  v8f acc[2][2] = {};

  for (int k0 = 0; k0 < Kdim; k0 += 32) {
    // ---------- stage A (uniform per-chunk segment select) ----------
    {
      __bf16* dst = &lds_a[ar][aks];
      if (AMODE == 0) {
        if (arow_ok && a_vec_ok && (k0 + 32 <= Kdim)) {
          stage16_vec(ap0 + k0 + aks, dst);
          if (k0 + 64 <= Kdim)
            __builtin_prefetch(ap0 + k0 + 32 + aks, 0, 1);   // global_prefetch next chunk
        } else if (arow_ok) {
          #pragma unroll
          for (int j = 0; j < 16; ++j) {
            int k = k0 + aks + j;
            dst[j] = (k < Kdim) ? (__bf16)ap0[k] : (__bf16)0.f;
          }
        } else stage16_zero(dst);
      } else {
        const float* src = nullptr; int kb = 0;
        if (k0 < A.s0)        { src = ap0; kb = k0; }
        else if (k0 < A.s01)  { src = ap1; kb = k0 - A.s0; }
        if (src) {
          if (arow_ok) stage16_vec(src + kb + aks, dst);
          else         stage16_zero(dst);
        } else {             // edge-feature tail chunk (AMODE 1 only)
          #pragma unroll
          for (int j = 0; j < 16; ++j) {
            int k = k0 + aks + j;
            dst[j] = (arow_ok && k < Kdim) ? (__bf16)ap2[k - A.s01] : (__bf16)0.f;
          }
        }
      }
    }
    // ---------- stage B (bf16 weights, coalesced across n) ----------
    {
      #pragma unroll
      for (int j = 0; j < 8; ++j) {
        int kk = bk0 + j;
        int k  = k0 + kk;
        lds_b[bn][kk] = (bcol_ok && k < Kdim) ? W[(size_t)k * N + bcol_g] : (__bf16)0.f;
      }
    }
    __syncthreads();

    // ---------- fragments (ISA 7.12.2 wave32 layouts) + 4 WMMAs ----------
    v16bf avf[2], bvf[2];
    #pragma unroll
    for (int i = 0; i < 2; ++i) {
      int arow = wm * 32 + i * 16 + (lane & 15);
      int ka   = (lane < 16) ? 0 : 8;
      BF16Frag fa;
      fa.lo = *(const uint4*)(&lds_a[arow][ka]);
      fa.hi = *(const uint4*)(&lds_a[arow][ka + 16]);
      avf[i] = __builtin_bit_cast(v16bf, fa);
      int bcol = wn * 32 + i * 16 + (lane & 15);
      int kb   = (lane < 16) ? 0 : 8;
      BF16Frag fbv;
      fbv.lo = *(const uint4*)(&lds_b[bcol][kb]);
      fbv.hi = *(const uint4*)(&lds_b[bcol][kb + 16]);
      bvf[i] = __builtin_bit_cast(v16bf, fbv);
    }
    #pragma unroll
    for (int i = 0; i < 2; ++i)
      #pragma unroll
      for (int j = 0; j < 2; ++j)
        acc[i][j] = __builtin_amdgcn_wmma_f32_16x16x32_bf16(
            false, avf[i], false, bvf[j], (short)0, acc[i][j], false, false);
    __syncthreads();
  }

  // ---------- epilogue ----------
  #pragma unroll
  for (int i = 0; i < 2; ++i) {
    #pragma unroll
    for (int j = 0; j < 2; ++j) {
      #pragma unroll
      for (int r = 0; r < 8; ++r) {
        int lrow = wm * 32 + i * 16 + ((lane < 16) ? r : (8 + r));
        int grow = rowBlk + lrow;
        int gcol = colBlk + wn * 32 + j * 16 + (lane & 15);
        if (grow < M && gcol < N) {
          float v = acc[i][j][r];
          if (bias) v += bias[gcol];
          if (act)  v = v / (1.f + expf(-v));                  // SiLU
          if (mode == 0)      C[(size_t)grow * N + gcol]  = v;
          else if (mode == 1) C[(size_t)grow * N + gcol] += v;
          else atomicAdd(&C[(size_t)scat[grow] * N + gcol], v);
        }
      }
    }
  }
}

// ---------------------------------------------------------------------------
__global__ void cvt_bf16_kernel(const float* __restrict__ src,
                                __bf16* __restrict__ dst, int n)
{
  int i = blockIdx.x * 256 + threadIdx.x;
  if (i < n) dst[i] = (__bf16)src[i];
}

__global__ void build_gn_kernel(const float* __restrict__ cond,
                                const float* __restrict__ tgt,
                                const float* __restrict__ forc,
                                const float* __restrict__ gpos,
                                float* __restrict__ gn)
{
  int i = blockIdx.x * 256 + threadIdx.x;
  if (i >= NGRID * GINF) return;
  int n = i / GINF, c = i % GINF;
  float v;
  if      (c < 64)  v = cond[(size_t)c * NGRID + n];
  else if (c < 96)  v = tgt[(size_t)(c - 64) * NGRID + n];
  else if (c < 104) v = forc[(size_t)(c - 96) * NGRID + n];
  else              v = gpos[(size_t)n * 5 + (c - 104)];
  gn[(size_t)n * GLD + c] = v;
}

__global__ void noise_kernel(const float* __restrict__ sig,
                             const float* __restrict__ w0, const float* __restrict__ b0,
                             const float* __restrict__ w1, const float* __restrict__ b1,
                             float* __restrict__ ne)
{
  __shared__ float feats[64];
  __shared__ float h[128];
  int t = threadIdx.x;  // 128 threads
  float ls = logf(sig[0]);
  if (t < 32) {
    float freq = (2.f * 3.14159265358979323846f / 16.f) * exp2f((float)t);
    float ang = ls * freq;
    feats[t]      = sinf(ang);
    feats[t + 32] = cosf(ang);
  }
  __syncthreads();
  float a = b0[t];
  for (int i = 0; i < 64; ++i) a += feats[i] * w0[i * 128 + t];
  h[t] = a / (1.f + expf(-a));
  __syncthreads();
  if (t < 16) {
    float o = b1[t];
    for (int i = 0; i < 128; ++i) o += h[i] * w1[i * 16 + t];
    ne[t] = o;
  }
}

__global__ void cond_ss_kernel(const float* __restrict__ ne,
                               const float* __restrict__ w,
                               const float* __restrict__ b,
                               float* __restrict__ ss)
{
  int j = blockIdx.x * 256 + threadIdx.x;
  if (j >= 512) return;
  float a = b[j];
  for (int i = 0; i < 16; ++i) a += ne[i] * w[i * 512 + j];
  ss[j] = a;
}

__global__ __launch_bounds__(256) void cond_ln_kernel(
    const float* __restrict__ x, const float* __restrict__ ss,
    const float* __restrict__ resid, float* __restrict__ out, int M)
{
  int wave = threadIdx.x >> 5, lane = threadIdx.x & 31;
  int row = blockIdx.x * 8 + wave;
  if (row >= M) return;
  const float* xr = x + (size_t)row * 256;
  float v[8], s = 0.f, s2 = 0.f;
  #pragma unroll
  for (int i = 0; i < 8; ++i) {
    v[i] = xr[lane + i * 32];
    s += v[i]; s2 += v[i] * v[i];
  }
  for (int off = 16; off; off >>= 1) {
    s  += __shfl_xor(s,  off, 32);
    s2 += __shfl_xor(s2, off, 32);
  }
  float mu  = s * (1.f / 256.f);
  float var = s2 * (1.f / 256.f) - mu * mu;
  float rstd = rsqrtf(var + 1e-5f);
  #pragma unroll
  for (int i = 0; i < 8; ++i) {
    int c = lane + i * 32;
    float o = (v[i] - mu) * rstd * (1.f + ss[c]) + ss[256 + c];
    if (resid) o += resid[(size_t)row * 256 + c];
    out[(size_t)row * 256 + c] = o;
  }
}

__global__ __launch_bounds__(256) void attn_kernel(
    const float* __restrict__ q, const float* __restrict__ k,
    const float* __restrict__ v, const int* __restrict__ nbrs,
    float* __restrict__ o)
{
  int wave = threadIdx.x >> 5, lane = threadIdx.x & 31;
  int id = blockIdx.x * 8 + wave;
  if (id >= NMESH * 4) return;
  int n = id >> 2, hd = id & 3;
  const float* qp = q + (size_t)n * 256 + hd * 64;
  int nb = nbrs[(size_t)n * 32 + lane];
  const float* kp = k + (size_t)nb * 256 + hd * 64;
  float s = 0.f;
  for (int d = 0; d < 64; ++d) s += qp[d] * kp[d];
  s *= 0.125f;
  float m = s;
  for (int off = 16; off; off >>= 1) m = fmaxf(m, __shfl_xor(m, off, 32));
  float e = expf(s - m);
  float sum = e;
  for (int off = 16; off; off >>= 1) sum += __shfl_xor(sum, off, 32);
  float wgt = e / sum;
  const float* vp = v + (size_t)nb * 256 + hd * 64;
  for (int d = 0; d < 64; ++d) {
    float val = wgt * vp[d];
    for (int off = 16; off; off >>= 1) val += __shfl_xor(val, off, 32);
    if (lane == (d & 31)) o[(size_t)n * 256 + hd * 64 + d] = val;
  }
}

__global__ void zero_kernel(float* __restrict__ p, int n) {
  int i = blockIdx.x * 256 + threadIdx.x;
  if (i < n) p[i] = 0.f;
}

__global__ void transpose_out_kernel(const float* __restrict__ src, float* __restrict__ dst) {
  int i = blockIdx.x * 256 + threadIdx.x;
  if (i >= NGRID * 32) return;
  int n = i >> 5, c = i & 31;
  dst[(size_t)c * NGRID + n] = src[i];
}

// ---------------------------------------------------------------------------
extern "C" void kernel_launch(void* const* d_in, const int* in_sizes, int n_in,
                              void* d_out, int out_size, void* d_ws, size_t ws_size,
                              hipStream_t stream)
{
  (void)in_sizes; (void)n_in; (void)out_size; (void)ws_size;
  const float* cond = (const float*)d_in[0];
  const float* forc = (const float*)d_in[1];
  const float* tgt  = (const float*)d_in[2];
  const float* sig  = (const float*)d_in[3];
  const float* gpos = (const float*)d_in[4];
  const float* mpos = (const float*)d_in[5];
  const float* g2me = (const float*)d_in[6];
  const float* m2ge = (const float*)d_in[7];
  const int* g2ms = (const int*)d_in[8];
  const int* g2mr = (const int*)d_in[9];
  const int* m2gs = (const int*)d_in[10];
  const int* m2gr = (const int*)d_in[11];
  const int* nbrs = (const int*)d_in[12];
  const float* P  = (const float*)d_in[13];
  float* out = (float*)d_out;
  float* ws  = (float*)d_ws;

  // ---- params: flat buffer in jax pytree (sorted-dict-key) order ----
  struct Lin { const float* w; const float* b; };
  size_t cur = 0;
  auto takeLin = [&](int din, int dout) {
    Lin l; l.b = P + cur; cur += dout; l.w = P + cur; cur += (size_t)din * dout; return l;
  };
  auto takeCln = [&]() {
    Lin c; c.b = P + cur; cur += 512; c.w = P + cur; cur += 16 * 512; return c;
  };
  Lin g2m_cln = takeCln();
  Lin g2m_e0 = takeLin(516, 256), g2m_e1 = takeLin(256, 256);
  Lin g2m_n0 = takeLin(512, 256), g2m_n1 = takeLin(256, 256);
  Lin ge0 = takeLin(GINF, 256), ge1 = takeLin(256, 256), ge2 = takeLin(256, 256);
  Lin hd0 = takeLin(256, 256), hd1 = takeLin(256, 256), hd2 = takeLin(256, 32);
  struct Layer { Lin f0, f1, ln1, ln2; const float *wk, *wo, *wq, *wv; } L[2];
  for (int i = 0; i < 2; ++i) {
    L[i].f0 = takeLin(256, 1024);
    L[i].f1 = takeLin(1024, 256);
    L[i].ln1 = takeCln();
    L[i].ln2 = takeCln();
    L[i].wk = P + cur; cur += 65536;
    L[i].wo = P + cur; cur += 65536;
    L[i].wq = P + cur; cur += 65536;
    L[i].wv = P + cur; cur += 65536;
  }
  Lin m2g_cln = takeCln();
  Lin m2g_e0 = takeLin(516, 256), m2g_e1 = takeLin(256, 256);
  Lin m2g_n0 = takeLin(512, 256), m2g_n1 = takeLin(256, 256);
  Lin me0 = takeLin(5, 256), me1 = takeLin(256, 256), me2 = takeLin(256, 256);
  Lin no0 = takeLin(64, 128), no1 = takeLin(128, 16);

  // ---- workspace ----
  size_t wc = 0;
  auto wbuf = [&](size_t n) { float* p = ws + wc; wc += n; return p; };
  float* gn   = wbuf((size_t)NGRID * GLD);
  float* gl   = wbuf((size_t)NGRID * 256);
  float* gt1  = wbuf((size_t)NGRID * 256);
  float* gt2  = wbuf((size_t)NGRID * 256);
  float* agg  = wbuf((size_t)NGRID * 256);
  float* ml   = wbuf((size_t)NMESH * 256);
  float* mh   = wbuf((size_t)NMESH * 256);
  float* mq   = wbuf((size_t)NMESH * 256);
  float* mk   = wbuf((size_t)NMESH * 256);
  float* mv   = wbuf((size_t)NMESH * 256);
  float* mo   = wbuf((size_t)NMESH * 256);
  float* mff  = wbuf((size_t)NMESH * 1024);
  float* etmp = wbuf((size_t)ECHUNK * 256);
  float* ghd  = wbuf((size_t)NGRID * 32);
  float* neb  = wbuf(16);
  float* ssb  = wbuf(512);
  // bf16 weight arena (~2.8M elems): lives in L2 (192MB) for the whole launch
  __bf16* arena = (__bf16*)wbuf(1600000);
  size_t acur = 0;
  auto cvtW = [&](const float* w, size_t n) {
    __bf16* dst = arena + acur; acur += (n + 7) & ~(size_t)7;
    cvt_bf16_kernel<<<(n + 255) / 256, 256, 0, stream>>>(w, dst, (int)n);
    return (const __bf16*)dst;
  };
  struct LinH { const __bf16* w; const float* b; };
  auto cvtLin = [&](Lin l, int din, int dout) {
    LinH h; h.b = l.b; h.w = cvtW(l.w, (size_t)din * dout); return h;
  };

  LinH g2m_e0h = cvtLin(g2m_e0, 516, 256), g2m_e1h = cvtLin(g2m_e1, 256, 256);
  LinH g2m_n0h = cvtLin(g2m_n0, 512, 256), g2m_n1h = cvtLin(g2m_n1, 256, 256);
  LinH ge0h = cvtLin(ge0, GINF, 256), ge1h = cvtLin(ge1, 256, 256), ge2h = cvtLin(ge2, 256, 256);
  LinH hd0h = cvtLin(hd0, 256, 256), hd1h = cvtLin(hd1, 256, 256), hd2h = cvtLin(hd2, 256, 32);
  struct LayerH { LinH f0, f1; const __bf16 *wk, *wo, *wq, *wv; } LH[2];
  for (int i = 0; i < 2; ++i) {
    LH[i].f0 = cvtLin(L[i].f0, 256, 1024);
    LH[i].f1 = cvtLin(L[i].f1, 1024, 256);
    LH[i].wk = cvtW(L[i].wk, 65536);
    LH[i].wo = cvtW(L[i].wo, 65536);
    LH[i].wq = cvtW(L[i].wq, 65536);
    LH[i].wv = cvtW(L[i].wv, 65536);
  }
  LinH m2g_e0h = cvtLin(m2g_e0, 516, 256), m2g_e1h = cvtLin(m2g_e1, 256, 256);
  LinH m2g_n0h = cvtLin(m2g_n0, 512, 256), m2g_n1h = cvtLin(m2g_n1, 256, 256);
  LinH me0h = cvtLin(me0, 5, 256), me1h = cvtLin(me1, 256, 256), me2h = cvtLin(me2, 256, 256);

  auto plainA = [](const float* src, int ld) {
    AGather a{}; a.src0 = src; a.idx0 = nullptr; a.ld0 = ld;
    a.s0 = 0x7fffffff; a.s01 = 0x7fffffff;
    a.src1 = nullptr; a.idx1 = nullptr; a.ld1 = 0; a.src2 = nullptr; a.ld2 = 0;
    return a;
  };
  auto launch = [&](int amode, AGather A, int M, int Kd, const __bf16* W,
                    const float* b, int N, float* C, const int* scat,
                    int mode, int act) {
    dim3 g((M + 127) / 128, (N + 63) / 64);
    if (amode == 0)      wmma_gemm<0><<<g, 256, 0, stream>>>(A, M, Kd, W, b, N, C, scat, mode, act);
    else if (amode == 1) wmma_gemm<1><<<g, 256, 0, stream>>>(A, M, Kd, W, b, N, C, scat, mode, act);
    else                 wmma_gemm<2><<<g, 256, 0, stream>>>(A, M, Kd, W, b, N, C, scat, mode, act);
  };
  auto gemm = [&](AGather A, int M, int Kd, LinH lin, int N, float* C, int mode, int act) {
    launch(0, A, M, Kd, lin.w, lin.b, N, C, nullptr, mode, act);
  };
  auto condln = [&](const float* x, const float* resid, float* o, int M) {
    cond_ln_kernel<<<(M + 7) / 8, 256, 0, stream>>>(x, ssb, resid, o, M);
  };

  // 1) grid node features + noise embedding
  build_gn_kernel<<<((size_t)NGRID * GINF + 255) / 256, 256, 0, stream>>>(cond, tgt, forc, gpos, gn);
  noise_kernel<<<1, 128, 0, stream>>>(sig, no0.w, no0.b, no1.w, no1.b, neb);

  // 2) grid embed
  gemm(plainA(gn, GLD),  NGRID, GINF, ge0h, 256, gt1, 0, 1);
  gemm(plainA(gt1, 256), NGRID, 256,  ge1h, 256, gt2, 0, 1);
  gemm(plainA(gt2, 256), NGRID, 256,  ge2h, 256, gl,  0, 0);

  // 3) mesh embed
  gemm(plainA(mpos, 5),  NMESH, 5,   me0h, 256, mh, 0, 1);
  gemm(plainA(mh, 256),  NMESH, 256, me1h, 256, mq, 0, 1);
  gemm(plainA(mq, 256),  NMESH, 256, me2h, 256, ml, 0, 0);

  // 4) g2m bipartite (edge MLP with fused gather + fused segment_sum)
  zero_kernel<<<((size_t)NMESH * 256 + 255) / 256, 256, 0, stream>>>(agg, NMESH * 256);
  for (int c0 = 0; c0 < NEDGE; c0 += ECHUNK) {
    AGather ea{};
    ea.src0 = gl;  ea.idx0 = g2ms + c0; ea.ld0 = 256; ea.s0 = 256;
    ea.src1 = ml;  ea.idx1 = g2mr + c0; ea.ld1 = 256; ea.s01 = 512;
    ea.src2 = g2me + (size_t)c0 * 4;    ea.ld2 = 4;
    launch(1, ea, ECHUNK, 516, g2m_e0h.w, g2m_e0h.b, 256, etmp, nullptr, 0, 1);
    launch(0, plainA(etmp, 256), ECHUNK, 256, g2m_e1h.w, g2m_e1h.b, 256, agg, g2mr + c0, 2, 0);
  }
  {
    AGather na{};
    na.src0 = ml;  na.idx0 = nullptr; na.ld0 = 256; na.s0 = 256;
    na.src1 = agg; na.idx1 = nullptr; na.ld1 = 256; na.s01 = 512;
    na.src2 = nullptr; na.ld2 = 0;
    launch(2, na, NMESH, 512, g2m_n0h.w, g2m_n0h.b, 256, mff, nullptr, 0, 1);
    gemm(plainA(mff, 256), NMESH, 256, g2m_n1h, 256, mo, 0, 0);
  }
  cond_ss_kernel<<<2, 256, 0, stream>>>(neb, g2m_cln.w, g2m_cln.b, ssb);
  condln(mo, ml, ml, NMESH);

  // 5) transformer layers
  for (int li = 0; li < 2; ++li) {
    cond_ss_kernel<<<2, 256, 0, stream>>>(neb, L[li].ln1.w, L[li].ln1.b, ssb);
    condln(ml, nullptr, mh, NMESH);
    launch(0, plainA(mh, 256), NMESH, 256, LH[li].wq, nullptr, 256, mq, nullptr, 0, 0);
    launch(0, plainA(mh, 256), NMESH, 256, LH[li].wk, nullptr, 256, mk, nullptr, 0, 0);
    launch(0, plainA(mh, 256), NMESH, 256, LH[li].wv, nullptr, 256, mv, nullptr, 0, 0);
    attn_kernel<<<(NMESH * 4 + 7) / 8, 256, 0, stream>>>(mq, mk, mv, nbrs, mo);
    launch(0, plainA(mo, 256), NMESH, 256, LH[li].wo, nullptr, 256, ml, nullptr, 1, 0);
    cond_ss_kernel<<<2, 256, 0, stream>>>(neb, L[li].ln2.w, L[li].ln2.b, ssb);
    condln(ml, nullptr, mh, NMESH);
    gemm(plainA(mh, 256),   NMESH, 256,  LH[li].f0, 1024, mff, 0, 1);
    gemm(plainA(mff, 1024), NMESH, 1024, LH[li].f1, 256,  ml,  1, 0);
  }

  // 6) m2g bipartite
  zero_kernel<<<((size_t)NGRID * 256 + 255) / 256, 256, 0, stream>>>(agg, NGRID * 256);
  for (int c0 = 0; c0 < NEDGE; c0 += ECHUNK) {
    AGather ea{};
    ea.src0 = ml;  ea.idx0 = m2gs + c0; ea.ld0 = 256; ea.s0 = 256;
    ea.src1 = gl;  ea.idx1 = m2gr + c0; ea.ld1 = 256; ea.s01 = 512;
    ea.src2 = m2ge + (size_t)c0 * 4;    ea.ld2 = 4;
    launch(1, ea, ECHUNK, 516, m2g_e0h.w, m2g_e0h.b, 256, etmp, nullptr, 0, 1);
    launch(0, plainA(etmp, 256), ECHUNK, 256, m2g_e1h.w, m2g_e1h.b, 256, agg, m2gr + c0, 2, 0);
  }
  {
    AGather na{};
    na.src0 = gl;  na.idx0 = nullptr; na.ld0 = 256; na.s0 = 256;
    na.src1 = agg; na.idx1 = nullptr; na.ld1 = 256; na.s01 = 512;
    na.src2 = nullptr; na.ld2 = 0;
    launch(2, na, NGRID, 512, m2g_n0h.w, m2g_n0h.b, 256, gt1, nullptr, 0, 1);
    gemm(plainA(gt1, 256), NGRID, 256, m2g_n1h, 256, gt2, 0, 0);
  }
  cond_ss_kernel<<<2, 256, 0, stream>>>(neb, m2g_cln.w, m2g_cln.b, ssb);
  condln(gt2, gl, gl, NGRID);

  // 7) head + CHW transpose
  gemm(plainA(gl, 256),  NGRID, 256, hd0h, 256, gt1, 0, 1);
  gemm(plainA(gt1, 256), NGRID, 256, hd1h, 256, gt2, 0, 1);
  gemm(plainA(gt2, 256), NGRID, 256, hd2h, 32,  ghd, 0, 0);
  transpose_out_kernel<<<((size_t)NGRID * 32 + 255) / 256, 256, 0, stream>>>(ghd, out);
}
